// PicoLlmAttention_33827162423422
// MI455X (gfx1250) — matrix operand order repaired and checked by reading, not compiled, not run on hardware
//
#include <hip/hip_runtime.h>
#include <hip/hip_bf16.h>
#include <stdint.h>

typedef _Float16 v16h __attribute__((ext_vector_type(16)));
typedef _Float16 v4h  __attribute__((ext_vector_type(4)));
typedef float    v8f  __attribute__((ext_vector_type(8)));

#define N_EMBD   1024
#define N_HEADS  16
#define N_KV     4
#define HEAD_D   64
#define KV_DIM   256   // N_KV * HEAD_D
#define RMS_EPS  1.1920929e-07f

static __device__ __forceinline__ v8f wmma16(const v16h& a, const v16h& b, const v8f& c) {
    // D = A(16x32 f16) * B(32x16 f16) + C(16x16 f32)
    return __builtin_amdgcn_wmma_f32_16x16x32_f16(false, a, false, b, (short)0, c, false, false);
}

union Frag16 { v16h v; uint4 q[2]; };

// ---------------------------------------------------------------------------
// Generic C[M,N] = A[M,K] @ W[N,K]^T   (A fp32 or f16 -> f16, W fp32 -> f16)
// Block: 256 threads = 8 waves. BM=128, BN=64, BK=32. Each wave: 16 rows x 64 cols.
// ---------------------------------------------------------------------------
template<int TA_IS_F16>
__global__ __launch_bounds__(256)
void gemm_wmma_nt(const void* __restrict__ Ap, const float* __restrict__ W,
                  float* __restrict__ C, int M, int N, int K)
{
    constexpr int BM = 128, BN = 64, BK = 32, LDT = 40; // 40-half stride: bank-conflict-free
    __shared__ __align__(16) _Float16 As[BM][LDT];
    __shared__ __align__(16) _Float16 Ws[BN][LDT];

    const int tid  = threadIdx.x;
    const int wave = tid >> 5;
    const int lane = tid & 31;
    const int ln   = lane & 15;
    const int hi   = lane >> 4;
    const int m0   = blockIdx.x * BM;
    const int n0   = blockIdx.y * BN;

    v8f acc[4];
    for (int t = 0; t < 4; ++t)
        for (int r = 0; r < 8; ++r) acc[t][r] = 0.f;

    for (int k0 = 0; k0 < K; k0 += BK) {
        // Stage A tile (convert to f16)
        for (int i = tid; i < BM * (BK / 4); i += 256) {
            int row = i >> 3, kc = i & 7;
            v4h hv;
            if (TA_IS_F16) {
                hv = *(const v4h*)((const _Float16*)Ap + (size_t)(m0 + row) * K + k0 + kc * 4);
            } else {
                float4 f = *(const float4*)((const float*)Ap + (size_t)(m0 + row) * K + k0 + kc * 4);
                hv = (v4h){(_Float16)f.x, (_Float16)f.y, (_Float16)f.z, (_Float16)f.w};
            }
            *(v4h*)&As[row][kc * 4] = hv;
        }
        // Stage W tile (convert to f16)
        for (int i = tid; i < BN * (BK / 4); i += 256) {
            int row = i >> 3, kc = i & 7;
            float4 f = *(const float4*)(W + (size_t)(n0 + row) * K + k0 + kc * 4);
            *(v4h*)&Ws[row][kc * 4] = (v4h){(_Float16)f.x, (_Float16)f.y, (_Float16)f.z, (_Float16)f.w};
        }
        // Prefetch next K-chunk of the streamed operands (global_prefetch_b8)
        if (k0 + BK < K) {
            __builtin_prefetch(W + (size_t)(n0 + (tid & 63)) * K + k0 + BK, 0, 1);
            if (!TA_IS_F16)
                __builtin_prefetch((const float*)Ap + (size_t)(m0 + (tid & 127)) * K + k0 + BK, 0, 1);
        }
        __syncthreads();

        // A fragment: row = wave*16 + ln; low lanes K {0..7,16..23}, high lanes +8
        Frag16 a;
        const int mrow = wave * 16 + ln;
        const int koff = hi ? 8 : 0;
        a.q[0] = *(const uint4*)&As[mrow][koff];
        a.q[1] = *(const uint4*)&As[mrow][16 + koff];

        for (int t = 0; t < 4; ++t) {
            Frag16 b;
            const int nrow = t * 16 + ln;
            const int bk   = hi ? 8 : 0;  // low lanes K {0..7,16..23}, high lanes +8
            b.q[0] = *(const uint4*)&Ws[nrow][bk];
            b.q[1] = *(const uint4*)&Ws[nrow][bk + 16];
            acc[t] = wmma16(a.v, b.v, acc[t]);
        }
        __syncthreads();
    }

    for (int t = 0; t < 4; ++t)
        for (int r = 0; r < 8; ++r) {
            int m = m0 + wave * 16 + r + hi * 8;
            int n = n0 + t * 16 + ln;
            C[(size_t)m * N + n] = acc[t][r];
        }
}

// ---------------------------------------------------------------------------
// Per-row prep: gate + ve add (V), RoPE + RMS-norm * 1.2 (Q,K); fp32 -> f16
// One wave per (b, s, row-type). Row types: 0..15 Q heads, 16..19 K, 20..23 V.
// Lane j owns dims (j, j+32) => natural RoPE pairing.
// ---------------------------------------------------------------------------
__global__ __launch_bounds__(256)
void prep_qkv(const float* __restrict__ x,   const float* __restrict__ ve,
              const float* __restrict__ cosb, const float* __restrict__ sinb,
              const float* __restrict__ Qf,  const float* __restrict__ Kf,
              const float* __restrict__ Vf,  const float* __restrict__ Wgate,
              _Float16* __restrict__ Qh, _Float16* __restrict__ Kh,
              _Float16* __restrict__ Vh, int B, int S)
{
    const int wave = threadIdx.x >> 5;
    const int lane = threadIdx.x & 31;
    const long row = (long)blockIdx.x * 8 + wave;
    const long rowsPerB = (long)S * 24;
    const int b = (int)(row / rowsPerB);
    const long rem = row - (long)b * rowsPerB;
    const int s = (int)(rem / 24);
    const int t = (int)(rem % 24);

    if (t < 20) {
        const float* src; _Float16* dst; int head, stride, nh;
        if (t < 16) { src = Qf; head = t;      stride = N_EMBD; dst = Qh; nh = N_HEADS; }
        else        { src = Kf; head = t - 16; stride = KV_DIM; dst = Kh; nh = N_KV;    }
        const float c  = cosb[(size_t)s * 32 + lane];
        const float sn = sinb[(size_t)s * 32 + lane];
        const size_t base = ((size_t)b * S + s) * stride + head * 64;
        const float t1 = src[base + lane];
        const float t2 = src[base + 32 + lane];
        const float o1 =  t1 * c  + t2 * sn;
        const float o2 = -t1 * sn + t2 * c;
        float ss = o1 * o1 + o2 * o2;
        ss += __shfl_xor(ss, 1);  ss += __shfl_xor(ss, 2);
        ss += __shfl_xor(ss, 4);  ss += __shfl_xor(ss, 8);
        ss += __shfl_xor(ss, 16);
        const float rs = rsqrtf(ss * (1.f / 64.f) + RMS_EPS) * 1.2f;
        const size_t ob = ((size_t)(b * nh + head) * S + s) * 64;
        dst[ob + lane]      = (_Float16)(o1 * rs);
        dst[ob + 32 + lane] = (_Float16)(o2 * rs);
    } else {
        const int kvh = t - 20;
        float dot = 0.f;
        const size_t xb = ((size_t)b * S + s) * N_EMBD;
        for (int cch = 0; cch < 12; ++cch) dot += x[xb + cch] * Wgate[kvh * 12 + cch];
        const float g = 3.f / (1.f + __expf(-dot));
        const size_t vb = ((size_t)b * S + s) * KV_DIM + kvh * 64;
        const float o1 = Vf[vb + lane]      + g * ve[vb + lane];
        const float o2 = Vf[vb + 32 + lane] + g * ve[vb + 32 + lane];
        const size_t ob = ((size_t)(b * N_KV + kvh) * S + s) * 64;
        Vh[ob + lane]      = (_Float16)o1;
        Vh[ob + 32 + lane] = (_Float16)o2;
    }
}

// ---------------------------------------------------------------------------
// Async stage of a 64x64 f16 tile (row stride 64 halves in global) into an
// LDS tile with 72-half stride, via GLOBAL_LOAD_ASYNC_TO_LDS_B128 (ASYNCcnt).
// LDS address operand = low 32 bits of the generic pointer (aperture rule).
// 512 x b128 chunks, 128 threads -> 4 issues per thread.
// ---------------------------------------------------------------------------
static __device__ __forceinline__ void async_tile_load_64x64(
    _Float16 (&dst)[64][72], const _Float16* __restrict__ src, int tid)
{
    for (int i = tid; i < 512; i += 128) {
        const int row = i >> 3, cch = i & 7;
        const unsigned lds_off = (unsigned)(size_t)&dst[row][cch * 8];
        const unsigned voff    = (unsigned)((row * 64 + cch * 8) * sizeof(_Float16));
        asm volatile("global_load_async_to_lds_b128 %0, %1, %2"
                     :: "v"(lds_off), "v"(voff), "s"(src)
                     : "memory");
    }
}

// ---------------------------------------------------------------------------
// Flash attention, sliding causal window. Block = 4 waves, 64-query tile.
// Scores and P@V both via v_wmma_f32_16x16x32_f16. GQA: head h -> kv head h/4.
// ---------------------------------------------------------------------------
__global__ __launch_bounds__(128)
void flash_attn(const _Float16* __restrict__ Qh, const _Float16* __restrict__ Kh,
                const _Float16* __restrict__ Vh, const int* __restrict__ amask,
                const int* __restrict__ winp, _Float16* __restrict__ Yh,
                int B, int S)
{
    constexpr int LDT = 72; // 72-half stride: 16B aligned, conflict-free
    __shared__ __align__(16) _Float16 Qs[64][LDT];
    __shared__ __align__(16) _Float16 Ks[64][LDT];
    __shared__ __align__(16) _Float16 Vt[64][LDT];      // transposed: [dim][key]
    __shared__ __align__(16) _Float16 Ps[4][16][LDT];   // per-wave P tile

    const int tid  = threadIdx.x;
    const int wave = tid >> 5;
    const int lane = tid & 31;
    const int ln   = lane & 15;
    const int hi   = lane >> 4;
    const int q0   = blockIdx.x * 64;
    const int bh   = blockIdx.y;
    const int b    = bh >> 4;
    const int h    = bh & 15;
    const int kvh  = h >> 2;
    const int window = winp[0];

    const _Float16* Qg = Qh + ((size_t)(b * N_HEADS + h) * S + q0) * 64;
    const _Float16* Kg = Kh + ((size_t)(b * N_KV + kvh) * S) * 64;
    const _Float16* Vg = Vh + ((size_t)(b * N_KV + kvh) * S) * 64;

    // Stage Q tile once (async DMA into LDS)
    async_tile_load_64x64(Qs, Qg, tid);
    asm volatile("s_wait_asynccnt 0x0" ::: "memory");
    __syncthreads();

    // Hoist Q A-fragments (dims 0..31 and 32..63)
    Frag16 a0, a1;
    {
        const int mrow = wave * 16 + ln;
        const int ko = hi ? 8 : 0;
        a0.q[0] = *(const uint4*)&Qs[mrow][ko];
        a0.q[1] = *(const uint4*)&Qs[mrow][16 + ko];
        a1.q[0] = *(const uint4*)&Qs[mrow][32 + ko];
        a1.q[1] = *(const uint4*)&Qs[mrow][48 + ko];
    }

    v8f O[4];
    float mrun[8], lrun[8];
    for (int t = 0; t < 4; ++t) for (int r = 0; r < 8; ++r) O[t][r] = 0.f;
    for (int r = 0; r < 8; ++r) { mrun[r] = -1e30f; lrun[r] = 0.f; }

    const int ktEnd = q0 >> 6;
    int lo = q0 - window + 1; if (lo < 0) lo = 0;
    const int ktStart = lo >> 6;
    const float scale = 0.125f; // 1/sqrt(64)

    for (int kt = ktStart; kt <= ktEnd; ++kt) {
        const int k0 = kt * 64;
        __syncthreads(); // previous tile fully consumed
        // K tile: async DMA straight to LDS (no VGPR staging)
        async_tile_load_64x64(Ks, Kg + (size_t)k0 * 64, tid);
        // V tile: staged transposed [dim][key] (scalar path, does the transpose)
        for (int i = tid; i < 64 * 64; i += 128) {
            int key = i >> 6, d = i & 63;
            Vt[d][key] = Vg[(size_t)(k0 + key) * 64 + d];
        }
        asm volatile("s_wait_asynccnt 0x0" ::: "memory");
        __syncthreads();

        // ---- scores: S = Q * K^T ----
        v8f sc[4];
        for (int t = 0; t < 4; ++t) {
            Frag16 bf;
            const int krow = t * 16 + ln;
            const int bk = hi ? 8 : 0;
            bf.q[0] = *(const uint4*)&Ks[krow][bk];
            bf.q[1] = *(const uint4*)&Ks[krow][bk + 16];
            v8f z; for (int r = 0; r < 8; ++r) z[r] = 0.f;
            sc[t] = wmma16(a0.v, bf.v, z);
            bf.q[0] = *(const uint4*)&Ks[krow][32 + bk];
            bf.q[1] = *(const uint4*)&Ks[krow][32 + bk + 16];
            sc[t] = wmma16(a1.v, bf.v, sc[t]);
        }

        // ---- mask + scale ----
        int am[4];
        for (int t = 0; t < 4; ++t) am[t] = amask[(size_t)b * S + k0 + t * 16 + ln];
        float mloc[8];
        for (int r = 0; r < 8; ++r) {
            const int q = q0 + wave * 16 + r + hi * 8;
            float mm = -1e30f;
            for (int t = 0; t < 4; ++t) {
                const int k = k0 + t * 16 + ln;
                const bool ok = (k <= q) && ((q - k) < window) && (am[t] != 0);
                const float v = ok ? sc[t][r] * scale : -1e30f;
                sc[t][r] = v;
                mm = fmaxf(mm, v);
            }
            mloc[r] = mm;
        }
        for (int r = 0; r < 8; ++r) { // row max across the 16 lanes holding that row
            float mm = mloc[r];
            mm = fmaxf(mm, __shfl_xor(mm, 1));
            mm = fmaxf(mm, __shfl_xor(mm, 2));
            mm = fmaxf(mm, __shfl_xor(mm, 4));
            mm = fmaxf(mm, __shfl_xor(mm, 8));
            mloc[r] = mm;
        }

        // ---- online softmax; emit P (f16) to per-wave LDS ----
        for (int r = 0; r < 8; ++r) {
            const float mnew = fmaxf(mrun[r], mloc[r]);
            const float f = __expf(mrun[r] - mnew);
            mrun[r] = mnew;
            float psum = 0.f;
            for (int t = 0; t < 4; ++t) {
                const float sv = sc[t][r];
                const float p = (sv > -1e29f) ? __expf(sv - mnew) : 0.f;
                psum += p;
                Ps[wave][r + hi * 8][t * 16 + ln] = (_Float16)p;
            }
            psum += __shfl_xor(psum, 1);
            psum += __shfl_xor(psum, 2);
            psum += __shfl_xor(psum, 4);
            psum += __shfl_xor(psum, 8);
            lrun[r] = lrun[r] * f + psum;
            for (int t = 0; t < 4; ++t) O[t][r] *= f;
        }
        asm volatile("s_wait_dscnt 0x0" ::: "memory"); // P writes visible to our own reads

        // ---- O += P * V ----
        for (int cch = 0; cch < 2; ++cch) { // key chunks of 32
            Frag16 ap;
            const int ko = cch * 32 + (hi ? 8 : 0);
            ap.q[0] = *(const uint4*)&Ps[wave][ln][ko];
            ap.q[1] = *(const uint4*)&Ps[wave][ln][ko + 16];
            for (int t = 0; t < 4; ++t) {
                Frag16 bv;
                const int drow = t * 16 + ln;
                const int bk = cch * 32 + (hi ? 8 : 0);
                bv.q[0] = *(const uint4*)&Vt[drow][bk];
                bv.q[1] = *(const uint4*)&Vt[drow][bk + 16];
                O[t] = wmma16(ap.v, bv.v, O[t]);
            }
        }
    }

    // ---- epilogue: y = O / l, layout [b][s][h*64+d] for the proj GEMM ----
    for (int r = 0; r < 8; ++r) {
        const int q = q0 + wave * 16 + r + hi * 8;
        const float inv = lrun[r] > 0.f ? 1.f / lrun[r] : 0.f;
        const size_t ob = ((size_t)b * S + q) * (size_t)N_EMBD + h * 64;
        for (int t = 0; t < 4; ++t)
            Yh[ob + t * 16 + ln] = (_Float16)(O[t][r] * inv);
    }
}

// ---------------------------------------------------------------------------
extern "C" void kernel_launch(void* const* d_in, const int* in_sizes, int n_in,
                              void* d_out, int out_size, void* d_ws, size_t ws_size,
                              hipStream_t stream)
{
    const float* x     = (const float*)d_in[0];
    const float* ve    = (const float*)d_in[1];
    const float* cosb  = (const float*)d_in[2];
    const float* sinb  = (const float*)d_in[3];
    const float* Wq    = (const float*)d_in[4];
    const float* Wk    = (const float*)d_in[5];
    const float* Wv    = (const float*)d_in[6];
    const float* Wproj = (const float*)d_in[7];
    const float* Wgate = (const float*)d_in[8];
    const int*   amask = (const int*)d_in[9];
    const int*   winp  = (const int*)d_in[10];

    const int BS = in_sizes[0] / N_EMBD;  // B*S = 4096
    const int S  = in_sizes[2] / 32;      // cos: (S,1,32)
    const int B  = BS / S;

    char* ws = (char*)d_ws;
    float*    Qf = (float*)ws;     ws += (size_t)BS * N_EMBD * 4;
    float*    Kf = (float*)ws;     ws += (size_t)BS * KV_DIM * 4;
    float*    Vf = (float*)ws;     ws += (size_t)BS * KV_DIM * 4;
    _Float16* Qh = (_Float16*)ws;  ws += (size_t)BS * N_EMBD * 2;
    _Float16* Kh = (_Float16*)ws;  ws += (size_t)BS * KV_DIM * 2;
    _Float16* Vh = (_Float16*)ws;  ws += (size_t)BS * KV_DIM * 2;
    _Float16* Yh = (_Float16*)ws;  ws += (size_t)BS * N_EMBD * 2;

    const dim3 blk256(256);
    // QKV projections (fp32 A -> f16 WMMA)
    gemm_wmma_nt<0><<<dim3(BS / 128, N_EMBD / 64), blk256, 0, stream>>>(x, Wq, Qf, BS, N_EMBD, N_EMBD);
    gemm_wmma_nt<0><<<dim3(BS / 128, KV_DIM / 64), blk256, 0, stream>>>(x, Wk, Kf, BS, KV_DIM, N_EMBD);
    gemm_wmma_nt<0><<<dim3(BS / 128, KV_DIM / 64), blk256, 0, stream>>>(x, Wv, Vf, BS, KV_DIM, N_EMBD);
    // gate + RoPE + RMS-norm -> f16 per-head layout
    prep_qkv<<<dim3((unsigned)((long)BS * 24 / 8)), blk256, 0, stream>>>(
        x, ve, cosb, sinb, Qf, Kf, Vf, Wgate, Qh, Kh, Vh, B, S);
    // windowed flash attention
    flash_attn<<<dim3(S / 64, B * N_HEADS), dim3(128), 0, stream>>>(
        Qh, Kh, Vh, amask, winp, Yh, B, S);
    // output projection (f16 A -> fp32 out)
    gemm_wmma_nt<1><<<dim3(BS / 128, N_EMBD / 64), blk256, 0, stream>>>(
        Yh, Wproj, (float*)d_out, BS, N_EMBD, N_EMBD);
}
